// SimpleRNN_87308095193886
// MI455X (gfx1250) — matrix operand order repaired and checked by reading, not compiled, not run on hardware
//
#include <hip/hip_runtime.h>
#include <math.h>

// ---------------------------------------------------------------------------
// SimpleRNN for MI455X (gfx1250, wave32, WMMA).
//   phase 1: xw = x @ W_xh + b_h          (WMMA f16 GEMM, f32 accum/out)
//   phase 2: sequential scan, batch-split into 4 independent workgroups,
//            h double-buffered in LDS (f16, padded rows), W_hh streamed
//            from the 192MB L2 as f16, hardware v_tanh_f32, history written
//            with global_store_async_from_lds_b128 (ASYNCcnt path)
//   phase 3: logits = hs @ W_o + b_o      (WMMA f16 GEMM, hs stored f16)
// ---------------------------------------------------------------------------

typedef __attribute__((ext_vector_type(16))) _Float16 v16h;
typedef __attribute__((ext_vector_type(8)))  _Float16 v8h;
typedef __attribute__((ext_vector_type(8)))  float    v8f;

#define DD     512             // D_X == D_H == D_OUT
#define BATCH  64
#define SEQ    1024
#define MTOT   (BATCH * SEQ)   // 65536
#define HPAD   520             // LDS row stride (f16): 1040B = 260 dwords ->
                               // consecutive rows shift 4 banks (A-frag reads)
#define BUFSZ  (16 * HPAD)     // one h buffer (f16 elems)

// hardware tanh (gfx1250 TRANS op); v_nop covers the TRANS RAW hazard the
// compiler cannot see through inline asm.
__device__ __forceinline__ float htanh(float x) {
    float y;
    asm volatile("v_tanh_f32 %0, %1\n\tv_nop" : "=v"(y) : "v"(x));
    return y;
}

// raw LDS byte offset of a __shared__ object (generic -> addrspace(3))
typedef __attribute__((address_space(3))) const void* lds_cvp;
__device__ __forceinline__ unsigned ldsOff(const void* p) {
    return (unsigned)(unsigned long long)(lds_cvp)p;
}

// async DMA: 16 bytes LDS -> global, tracked by ASYNCcnt
__device__ __forceinline__ void asyncStoreB128(void* gptr, unsigned ldsoff) {
    unsigned long long ga = (unsigned long long)gptr;
    asm volatile("global_store_async_from_lds_b128 %0, %1, off"
                 :: "v"(ga), "v"(ldsoff) : "memory");
}

// ------------------------- f32 -> f16 weight convert -----------------------
__global__ void cvt_f32_f16(const float* __restrict__ src,
                            _Float16* __restrict__ dst, int n) {
    int i = blockIdx.x * blockDim.x + threadIdx.x;
    if (i < n) dst[i] = (_Float16)src[i];
}

// ------------------------- WMMA GEMM:  out = A*B + bias --------------------
// A: [M][K] (f32 if !A_IS_HALF, else f16), row-major
// B: [K][N] f16 row-major (lane holds one B column, rows kb+8*half+j and kb+16+8*half+j)
// out: [M][N] f32.  Block tile 128x256, 8 waves, wave tile 64x64.
// grid: x = N-block (so both N-halves of one row slab reuse A via L2),
//       y = M-block.
template <bool A_IS_HALF>
__global__ void __launch_bounds__(256)
gemm_bias(const void* __restrict__ Aptr, const _Float16* __restrict__ Bw,
          const float* __restrict__ bias, float* __restrict__ out,
          int M, int K, int N) {
    const int lane  = threadIdx.x & 31;
    const int wave  = threadIdx.x >> 5;
    const int half  = lane >> 4;
    const int l16   = lane & 15;
    const int rowBase = blockIdx.y * 128 + (wave & 1) * 64;
    const int colBase = blockIdx.x * 256 + (wave >> 1) * 64;

    v8f c[4][4] = {};

    for (int kb = 0; kb < K; kb += 32) {
        v16h a[4], b[4];
#pragma unroll
        for (int mt = 0; mt < 4; ++mt) {
            const size_t rbase = (size_t)(rowBase + mt * 16 + l16) * K + kb + 8 * half;
            if (A_IS_HALF) {
                const _Float16* A = (const _Float16*)Aptr;
                v8h lo = *(const v8h*)(A + rbase);
                v8h hi = *(const v8h*)(A + rbase + 16);
#pragma unroll
                for (int j = 0; j < 8; ++j) { a[mt][j] = lo[j]; a[mt][8 + j] = hi[j]; }
            } else {
                const float* A = (const float*)Aptr;
                v8f lo = *(const v8f*)(A + rbase);
                v8f hi = *(const v8f*)(A + rbase + 16);
#pragma unroll
                for (int j = 0; j < 8; ++j) {
                    a[mt][j]     = (_Float16)lo[j];
                    a[mt][8 + j] = (_Float16)hi[j];
                }
            }
        }
#pragma unroll
        for (int nt = 0; nt < 4; ++nt)
            for (int j = 0; j < 8; ++j) {
                b[nt][j]     = Bw[(size_t)(kb + 8 * half + j) * N + colBase + nt * 16 + l16];
                b[nt][8 + j] = Bw[(size_t)(kb + 16 + 8 * half + j) * N + colBase + nt * 16 + l16];
            }

#pragma unroll
        for (int mt = 0; mt < 4; ++mt)
#pragma unroll
            for (int nt = 0; nt < 4; ++nt)
                c[mt][nt] = __builtin_amdgcn_wmma_f32_16x16x32_f16(
                    false, a[mt], false, b[nt], (short)0, c[mt][nt], false, false);
    }

#pragma unroll
    for (int nt = 0; nt < 4; ++nt) {
        const int col = colBase + nt * 16 + l16;
        const float bv = bias[col];
#pragma unroll
        for (int mt = 0; mt < 4; ++mt) {
            const int row0 = rowBase + mt * 16 + 8 * half;
#pragma unroll
            for (int i = 0; i < 8; ++i)
                out[(size_t)(row0 + i) * N + col] = c[mt][nt][i] + bv;
        }
    }
}

// ------------------------- sequential RNN scan -----------------------------
// 4 blocks (independent batch groups of 16 rows), 512 threads = 16 waves.
// Wave w owns output columns [32w, 32w+32).  h double-buffered in LDS.
__global__ void __launch_bounds__(512)
rnn_scan(const float* __restrict__ xw,      // [B*T][512] f32 (incl. b_h)
         const _Float16* __restrict__ Whh,  // [512][512] f16 row-major
         _Float16* __restrict__ hs,         // [B*T][512] f16 out
         float* __restrict__ h_last)        // [64][512] f32 out
{
    __shared__ _Float16 hsm[2 * BUFSZ];
    const int tid  = threadIdx.x;
    const int lane = tid & 31;
    const int wave = tid >> 5;        // 0..15
    const int half = lane >> 4;
    const int l16  = lane & 15;
    const int bg   = blockIdx.x;      // batch rows [bg*16, bg*16+16)
    const int n0   = wave * 32;

    // per-thread async-copy chunk: row r (0..15), 32B column slice c32
    const int ar   = tid >> 5;
    const int ac   = (tid & 31) * 32;                 // byte offset in row
    const unsigned ldsRow0 = ldsOff(hsm) + (unsigned)(ar * HPAD * 2 + ac);
    char* hsRowG = (char*)hs + ((size_t)(bg * 16 + ar) * SEQ) * (DD * 2) + ac;

    for (int i = tid; i < 2 * BUFSZ; i += blockDim.x) hsm[i] = (_Float16)0.0f;
    __syncthreads();

    // software-pipelined xw registers for step t
    float nx0[8], nx1[8];
#pragma unroll
    for (int i = 0; i < 8; ++i) {
        const size_t base = ((size_t)(bg * 16 + i + 8 * half) * SEQ) * DD;
        nx0[i] = xw[base + n0 + l16];
        nx1[i] = xw[base + n0 + 16 + l16];
    }

    int p = 0;
    for (int t = 0; t < SEQ; ++t) {
        const int q = p ^ 1;
        const _Float16* hr = hsm + p * BUFSZ;   // read buffer
        _Float16*       hw = hsm + q * BUFSZ;   // write buffer

        v8f c0, c1;
#pragma unroll
        for (int i = 0; i < 8; ++i) { c0[i] = nx0[i]; c1[i] = nx1[i]; }

        // prefetch xw for t+1 (overlaps k-loop / tanh / barrier)
        if (t + 1 < SEQ) {
#pragma unroll
            for (int i = 0; i < 8; ++i) {
                const size_t base = ((size_t)(bg * 16 + i + 8 * half) * SEQ + (t + 1)) * DD;
                nx0[i] = xw[base + n0 + l16];
                nx1[i] = xw[base + n0 + 16 + l16];
            }
        }

        // h @ W_hh : A from LDS, B from L2-resident W_hh
#pragma unroll 4
        for (int kb = 0; kb < DD; kb += 32) {
            v16h a;
            {
                const int k0 = kb + 8 * half;
                v8h lo = *(const v8h*)(hr + l16 * HPAD + k0);
                v8h hi = *(const v8h*)(hr + l16 * HPAD + k0 + 16);
#pragma unroll
                for (int j = 0; j < 8; ++j) { a[j] = lo[j]; a[8 + j] = hi[j]; }
            }
            v16h b0, b1;
#pragma unroll
            for (int j = 0; j < 8; ++j) {
                b0[j]     = Whh[(size_t)(kb + 8 * half + j) * DD + n0 + l16];
                b0[8 + j] = Whh[(size_t)(kb + 16 + 8 * half + j) * DD + n0 + l16];
                b1[j]     = Whh[(size_t)(kb + 8 * half + j) * DD + n0 + 16 + l16];
                b1[8 + j] = Whh[(size_t)(kb + 16 + 8 * half + j) * DD + n0 + 16 + l16];
            }
            c0 = __builtin_amdgcn_wmma_f32_16x16x32_f16(false, a, false, b0,
                                                        (short)0, c0, false, false);
            c1 = __builtin_amdgcn_wmma_f32_16x16x32_f16(false, a, false, b1,
                                                        (short)0, c1, false, false);
        }

        float hn0[8], hn1[8];
#pragma unroll
        for (int i = 0; i < 8; ++i) { hn0[i] = htanh(c0[i]); hn1[i] = htanh(c1[i]); }

        // async stores issued at step t-2 read buffer q; drain them before
        // overwriting it (2 issued per step per thread, in-order completion)
        asm volatile("s_wait_asynccnt 0x2" ::: "memory");

#pragma unroll
        for (int i = 0; i < 8; ++i) {
            const int r = i + 8 * half;
            hw[r * HPAD + n0 + l16]      = (_Float16)hn0[i];
            hw[r * HPAD + n0 + 16 + l16] = (_Float16)hn1[i];
        }
        if (t == SEQ - 1) {
#pragma unroll
            for (int i = 0; i < 8; ++i) {
                const int b = bg * 16 + i + 8 * half;
                h_last[(size_t)b * DD + n0 + l16]      = hn0[i];
                h_last[(size_t)b * DD + n0 + 16 + l16] = hn1[i];
            }
        }
        __syncthreads();   // h_new visible (to waves and to the async engine)

        // DMA the fresh h tile (buffer q) to the hs history, 32B per thread
        {
            const unsigned lsrc = ldsRow0 + (unsigned)(q * BUFSZ * 2);
            char* g = hsRowG + (size_t)t * (DD * 2);
            asyncStoreB128(g,      lsrc);
            asyncStoreB128(g + 16, lsrc + 16);
        }
        p = q;
    }
}

// ------------------------------ launch -------------------------------------
extern "C" void kernel_launch(void* const* d_in, const int* in_sizes, int n_in,
                              void* d_out, int out_size, void* d_ws, size_t ws_size,
                              hipStream_t stream) {
    const float* x   = (const float*)d_in[0];   // [64,1024,512]
    const float* Wxh = (const float*)d_in[1];   // [512,512]
    const float* bh  = (const float*)d_in[2];   // [512]
    const float* Whh = (const float*)d_in[3];   // [512,512]
    const float* Wo  = (const float*)d_in[4];   // [512,512]
    const float* bo  = (const float*)d_in[5];   // [512]

    float* logits = (float*)d_out;                       // [64,1024,512]
    float* hlast  = logits + (size_t)MTOT * DD;          // [64,512]

    char* ws = (char*)d_ws;
    float*    xw    = (float*)ws;                                    // 134 MB
    _Float16* hs    = (_Float16*)(ws + (size_t)MTOT * DD * 4);       //  67 MB
    _Float16* Wxh16 = (_Float16*)(ws + (size_t)MTOT * DD * 4
                                     + (size_t)MTOT * DD * 2);
    _Float16* Whh16 = Wxh16 + DD * DD;
    _Float16* Wo16  = Whh16 + DD * DD;

    const int nW = DD * DD;
    cvt_f32_f16<<<(nW + 255) / 256, 256, 0, stream>>>(Wxh, Wxh16, nW);
    cvt_f32_f16<<<(nW + 255) / 256, 256, 0, stream>>>(Whh, Whh16, nW);
    cvt_f32_f16<<<(nW + 255) / 256, 256, 0, stream>>>(Wo,  Wo16,  nW);

    dim3 gg(DD / 256, MTOT / 128);
    gemm_bias<false><<<gg, 256, 0, stream>>>(x, Wxh16, bh, xw, MTOT, DD, DD);

    rnn_scan<<<4, 512, 0, stream>>>(xw, Whh16, hs, hlast);

    gemm_bias<true><<<gg, 256, 0, stream>>>(hs, Wo16, bo, logits, MTOT, DD, DD);
}
